// UpperBlock_73486890434739
// MI455X (gfx1250) — compile-verified, matrix operand order checked by reading, not run on hardware
//
#include <hip/hip_runtime.h>
#include <hip/hip_bf16.h>

// ---------------- problem constants ----------------
#define BB   8
#define SS   2048
#define EE   512
#define HH   8
#define DHD  64
#define FFD  2048
#define WIN  16
#define MTOT (BB * SS)          // 16384 rows
#define LD3E (3 * EE)           // qkv row stride

typedef __attribute__((ext_vector_type(16))) _Float16 v16h;
typedef __attribute__((ext_vector_type(8)))  _Float16 v8h;
typedef __attribute__((ext_vector_type(8)))  float    v8f;
typedef int v4i_vs __attribute__((vector_size(4 * sizeof(int))));

#define AS1 __attribute__((address_space(1)))
#define AS3 __attribute__((address_space(3)))

// ---------------- WMMA helpers ----------------
__device__ __forceinline__ v8f wmma32(v16h a, v16h b, v8f c) {
    // D = A(16x32 f16) * B(32x16 f16) + C(16x16 f32)
    return __builtin_amdgcn_wmma_f32_16x16x32_f16(
        /*neg_a=*/false, a, /*neg_b=*/false, b,
        /*c_mod=*/(short)0, c, /*reuse_a=*/false, /*reuse_b=*/false);
}

// A-fragment (16x32, rows) and B-fragment (32x16, cols) share this layout:
// lane L holds row/col (L&15); K values kb..kb+7 and kb+16..kb+23, kb = 8*(L>>4).
// Matrix stored row-major [R, ld] with the 32 K-values contiguous -> two b128 loads.
__device__ __forceinline__ v16h ld_frag(const _Float16* __restrict__ base, int ld,
                                        int rbase, int k0, int lane) {
    int r  = rbase + (lane & 15);
    int kb = k0 + ((lane >> 4) << 3);
    const _Float16* p = base + (size_t)r * ld + kb;
    v16h f;
    ((v8h*)&f)[0] = *(const v8h*)(p);
    ((v8h*)&f)[1] = *(const v8h*)(p + 16);
    return f;
}

__device__ __forceinline__ v16h ld_frag_clamp(const _Float16* __restrict__ base, int ld,
                                              int rbase, int k0, int lane, int rmax) {
    int r  = rbase + (lane & 15);
    if (r > rmax) r = rmax;
    int kb = k0 + ((lane >> 4) << 3);
    const _Float16* p = base + (size_t)r * ld + kb;
    v16h f;
    ((v8h*)&f)[0] = *(const v8h*)(p);
    ((v8h*)&f)[1] = *(const v8h*)(p + 16);
    return f;
}

__device__ __forceinline__ float gelu_f(float x) {
    const float c = 0.7978845608028654f;  // sqrt(2/pi)
    return 0.5f * x * (1.0f + tanhf(c * (x + 0.044715f * x * x * x)));
}

// ---------------- fp32 -> f16 convert ----------------
__global__ void cvt_f32_f16_kernel(const float* __restrict__ in,
                                   _Float16* __restrict__ out, int n) {
    int i = blockIdx.x * blockDim.x + threadIdx.x;
    if (i < n) out[i] = (_Float16)in[i];
}

// ---------------- LayerNorm (one wave per row of 512) ----------------
__global__ void ln_kernel(const float* __restrict__ x,
                          const float* __restrict__ w,
                          const float* __restrict__ b,
                          _Float16* __restrict__ out) {
    int lane = threadIdx.x & 31;
    int row  = blockIdx.x * (blockDim.x >> 5) + (threadIdx.x >> 5);
    const float* xr = x + (size_t)row * EE;
    float v[16];
    float s = 0.0f;
#pragma unroll
    for (int i = 0; i < 16; ++i) { v[i] = xr[lane + i * 32]; s += v[i]; }
#pragma unroll
    for (int m = 16; m >= 1; m >>= 1) s += __shfl_xor(s, m, 32);
    float mu = s * (1.0f / (float)EE);
    float var = 0.0f;
#pragma unroll
    for (int i = 0; i < 16; ++i) { float d = v[i] - mu; var += d * d; }
#pragma unroll
    for (int m = 16; m >= 1; m >>= 1) var += __shfl_xor(var, m, 32);
    float rs = rsqrtf(var * (1.0f / (float)EE) + 1e-5f);
    _Float16* orow = out + (size_t)row * EE;
#pragma unroll
    for (int i = 0; i < 16; ++i) {
        int c = lane + i * 32;
        orow[c] = (_Float16)((v[i] - mu) * rs * w[c] + b[c]);
    }
}

// ---------------- block-cooperative WMMA GEMM ----------------
// out = A[M,K] * W[N,K]^T + bias. Block (8 waves) computes 256 rows x 64 cols;
// the shared 64x32 f16 B-tile is staged into LDS once per k-step via
// GLOBAL_LOAD_ASYNC_TO_LDS_B128 (ASYNCcnt-tracked), each wave runs 8 WMMAs/k-step.
// MODE 0: store f16; MODE 1: outF = res + acc + bias (f32); MODE 2: f16(gelu).
#define BROW_PAD 40   // 32 halves + 8 pad -> 20-dword row stride, conflict-free frags
template <int MODE>
__global__ void gemm_wmma_kernel(const _Float16* __restrict__ A,
                                 const _Float16* __restrict__ Wm,
                                 const float* __restrict__ bias,
                                 const float* __restrict__ res,
                                 float* __restrict__ outF,
                                 _Float16* __restrict__ outH,
                                 int N, int K) {
    __shared__ _Float16 bs[64][BROW_PAD];
    int tid  = threadIdx.x;
    int lane = tid & 31;
    int wave = tid >> 5;
    int m0 = blockIdx.x * 256 + wave * 32;   // 32 rows per wave
    int n0 = blockIdx.y * 64;                // 64 cols per block (shared B tile)

    // staging assignment: thread -> (row of 64, 16B chunk of 4)
    int srow = tid >> 2;
    int schk = (tid & 3) * 8;

    v8f acc[2][4] = {};
    for (int k0 = 0; k0 < K; k0 += 32) {
        // A fragments (global, per-wave rows)
        v16h a0 = ld_frag(A, K, m0,      k0, lane);
        v16h a1 = ld_frag(A, K, m0 + 16, k0, lane);

        __syncthreads();   // previous iteration's B reads complete
        {
            const _Float16* gsrc = Wm + (size_t)(n0 + srow) * K + k0 + schk;
            _Float16* ldst = &bs[srow][schk];
#if __has_builtin(__builtin_amdgcn_global_load_async_to_lds_b128)
            __builtin_amdgcn_global_load_async_to_lds_b128(
                (AS1 v4i_vs*)(AS1 void*)gsrc, (AS3 v4i_vs*)(AS3 void*)ldst, 0, 0);
            asm volatile("s_wait_asynccnt 0x0" ::: "memory");
#else
            *(v8h*)ldst = *(const v8h*)gsrc;
#endif
        }
        __syncthreads();   // B tile visible to all waves

#pragma unroll
        for (int t = 0; t < 4; ++t) {
            v16h bfr;
            {
                int r  = t * 16 + (lane & 15);
                int kb = (lane >> 4) << 3;
                ((v8h*)&bfr)[0] = *(const v8h*)&bs[r][kb];
                ((v8h*)&bfr)[1] = *(const v8h*)&bs[r][kb + 16];
            }
            acc[0][t] = wmma32(a0, bfr, acc[0][t]);
            acc[1][t] = wmma32(a1, bfr, acc[1][t]);
        }
    }

    int hi = lane >> 4, nn = lane & 15;
#pragma unroll
    for (int mi = 0; mi < 2; ++mi) {
#pragma unroll
        for (int t = 0; t < 4; ++t) {
            int n = n0 + t * 16 + nn;
            float bv = bias[n];
#pragma unroll
            for (int r = 0; r < 8; ++r) {
                int m = m0 + mi * 16 + r + hi * 8;
                size_t idx = (size_t)m * N + n;
                float val = acc[mi][t][r] + bv;
                if constexpr (MODE == 0)      outH[idx] = (_Float16)val;
                else if constexpr (MODE == 1) outF[idx] = res[idx] + val;
                else                          outH[idx] = (_Float16)gelu_f(val);
            }
        }
    }
}

// ---------------- sliding-window attention (wave per 16-row tile) ----------------
__global__ void attn_kernel(const _Float16* __restrict__ qkv,
                            _Float16* __restrict__ o) {
    __shared__ _Float16 lds[8][16][32];
    int lane = threadIdx.x & 31;
    int wave = threadIdx.x >> 5;
    int tile = blockIdx.x * 8 + wave;
    const int S16 = SS / 16;
    int b  = tile / (HH * S16);
    int r2 = tile % (HH * S16);
    int h  = r2 / S16;
    int i0 = (r2 % S16) * 16;

    const _Float16* qb = qkv + (size_t)b * SS * LD3E + h * DHD;
    const _Float16* kb = qb + EE;
    const _Float16* vb = qb + 2 * EE;

    // Q as A-fragments over K = DH (two k-steps of 32)
    v16h qa0 = ld_frag(qb, LD3E, i0, 0,  lane);
    v16h qa1 = ld_frag(qb, LD3E, i0, 32, lane);

    // scores: two 16x16 tiles for keys [i0, i0+16) and [i0+16, i0+32)
    v8f s[2] = {};
#pragma unroll
    for (int t = 0; t < 2; ++t) {
        int j0 = i0 + t * 16;
        v16h kf0 = ld_frag_clamp(kb, LD3E, j0, 0,  lane, SS - 1);
        v16h kf1 = ld_frag_clamp(kb, LD3E, j0, 32, lane, SS - 1);
        s[t] = wmma32(qa0, kf0, s[t]);
        s[t] = wmma32(qa1, kf1, s[t]);
    }

    // scale + sliding-window mask (allowed: i <= j < i+W, j < S)
    const float scale = 0.125f;  // 1/sqrt(64)
    int hi = lane >> 4, nn = lane & 15;
#pragma unroll
    for (int r = 0; r < 8; ++r) {
        int i = i0 + r + hi * 8;
#pragma unroll
        for (int t = 0; t < 2; ++t) {
            int j = i0 + t * 16 + nn;
            bool ok = (j >= i) && (j < i + WIN) && (j < SS);
            s[t][r] = ok ? s[t][r] * scale : -1e9f;
        }
    }

    // row-wise softmax: rows live across 16 lanes of a half-wave per C-VGPR
#pragma unroll
    for (int r = 0; r < 8; ++r) {
        float mx = fmaxf(s[0][r], s[1][r]);
#pragma unroll
        for (int m = 8; m >= 1; m >>= 1) mx = fmaxf(mx, __shfl_xor(mx, m, 32));
        float e0 = __expf(s[0][r] - mx);
        float e1 = __expf(s[1][r] - mx);
        float sum = e0 + e1;
#pragma unroll
        for (int m = 8; m >= 1; m >>= 1) sum += __shfl_xor(sum, m, 32);
        float inv = 1.0f / sum;
        s[0][r] = e0 * inv;
        s[1][r] = e1 * inv;
    }

    // C-layout -> A-fragment relayout through LDS
#pragma unroll
    for (int r = 0; r < 8; ++r) {
        int m = r + hi * 8;
        lds[wave][m][nn]      = (_Float16)s[0][r];
        lds[wave][m][nn + 16] = (_Float16)s[1][r];
    }
    __syncthreads();
    v16h pa;
    {
        int rr  = lane & 15;
        int kbs = (lane >> 4) << 3;
        const _Float16* p = &lds[wave][rr][kbs];
        ((v8h*)&pa)[0] = *(const v8h*)(p);
        ((v8h*)&pa)[1] = *(const v8h*)(p + 16);
    }

    // o(16x64) = P(16x32) x V(32x64): per 16-col dh slice, gather V^T B-fragment
    _Float16* ob = o + (size_t)b * SS * EE + h * DHD;
    int kbs = (lane >> 4) << 3;
#pragma unroll
    for (int t = 0; t < 4; ++t) {
        int col = t * 16 + nn;   // dh within head
        v16h vf;
#pragma unroll
        for (int u = 0; u < 8; ++u) {
            int j1 = i0 + kbs + u;       if (j1 > SS - 1) j1 = SS - 1;
            int j2 = i0 + kbs + 16 + u;  if (j2 > SS - 1) j2 = SS - 1;
            vf[u]     = vb[(size_t)j1 * LD3E + col];
            vf[8 + u] = vb[(size_t)j2 * LD3E + col];
        }
        v8f oc = {};
        oc = wmma32(pa, vf, oc);
#pragma unroll
        for (int r = 0; r < 8; ++r) {
            int m = r + hi * 8;
            ob[(size_t)(i0 + m) * EE + t * 16 + nn] = (_Float16)oc[r];
        }
    }
}

// ---------------- workspace layout (bytes) ----------------
#define OFF_WQKV ((size_t)0)                       // 3E*E f16   = 1,572,864
#define OFF_WO   (OFF_WQKV + (size_t)3*EE*EE*2)
#define OFF_W1   (OFF_WO   + (size_t)EE*EE*2)
#define OFF_W2   (OFF_W1   + (size_t)FFD*EE*2)
#define OFF_XN   (OFF_W2   + (size_t)EE*FFD*2)     // [M,E] f16
#define OFF_QKV  (OFF_XN   + (size_t)MTOT*EE*2)    // [M,3E] f16
#define OFF_O    (OFF_QKV  + (size_t)MTOT*3*EE*2)  // [M,E] f16
#define OFF_X2   (OFF_O    + (size_t)MTOT*EE*2)    // [M,E] f32
#define OFF_XN2  (OFF_X2   + (size_t)MTOT*EE*4)    // [M,E] f16
#define OFF_H    (OFF_XN2  + (size_t)MTOT*EE*2)    // [M,FF] f16

extern "C" void kernel_launch(void* const* d_in, const int* in_sizes, int n_in,
                              void* d_out, int out_size, void* d_ws, size_t ws_size,
                              hipStream_t stream) {
    const float* x     = (const float*)d_in[0];
    const float* ln1_w = (const float*)d_in[1];
    const float* ln1_b = (const float*)d_in[2];
    const float* ln2_w = (const float*)d_in[3];
    const float* ln2_b = (const float*)d_in[4];
    const float* w_qkv = (const float*)d_in[5];
    const float* b_qkv = (const float*)d_in[6];
    const float* w_o   = (const float*)d_in[7];
    const float* b_o   = (const float*)d_in[8];
    const float* w1    = (const float*)d_in[9];
    const float* b1    = (const float*)d_in[10];
    const float* w2    = (const float*)d_in[11];
    const float* b2    = (const float*)d_in[12];
    float* out = (float*)d_out;

    char* ws = (char*)d_ws;
    _Float16* wqkv_h = (_Float16*)(ws + OFF_WQKV);
    _Float16* wo_h   = (_Float16*)(ws + OFF_WO);
    _Float16* w1_h   = (_Float16*)(ws + OFF_W1);
    _Float16* w2_h   = (_Float16*)(ws + OFF_W2);
    _Float16* xn_h   = (_Float16*)(ws + OFF_XN);
    _Float16* qkv_h  = (_Float16*)(ws + OFF_QKV);
    _Float16* o_h    = (_Float16*)(ws + OFF_O);
    float*    x2     = (float*)   (ws + OFF_X2);
    _Float16* xn2_h  = (_Float16*)(ws + OFF_XN2);
    _Float16* h_h    = (_Float16*)(ws + OFF_H);

    dim3 blk(256);

    // 1) weight conversions fp32 -> f16
    {
        int n;
        n = 3 * EE * EE; cvt_f32_f16_kernel<<<(n + 255) / 256, blk, 0, stream>>>(w_qkv, wqkv_h, n);
        n = EE * EE;     cvt_f32_f16_kernel<<<(n + 255) / 256, blk, 0, stream>>>(w_o, wo_h, n);
        n = FFD * EE;    cvt_f32_f16_kernel<<<(n + 255) / 256, blk, 0, stream>>>(w1, w1_h, n);
        n = EE * FFD;    cvt_f32_f16_kernel<<<(n + 255) / 256, blk, 0, stream>>>(w2, w2_h, n);
    }

    // 2) LN1: x -> xn (f16)
    ln_kernel<<<MTOT / 8, blk, 0, stream>>>(x, ln1_w, ln1_b, xn_h);

    // 3) QKV GEMM: [M,512] x [1536,512]^T -> f16 [M,1536]
    gemm_wmma_kernel<0><<<dim3(MTOT / 256, (3 * EE) / 64), blk, 0, stream>>>(
        xn_h, wqkv_h, b_qkv, nullptr, nullptr, qkv_h, 3 * EE, EE);

    // 4) sliding-window attention -> o (f16 [M,E])
    attn_kernel<<<(BB * HH * (SS / 16)) / 8, blk, 0, stream>>>(qkv_h, o_h);

    // 5) O-proj + residual: x2 = x + o*Wo^T + b_o  (f32)
    gemm_wmma_kernel<1><<<dim3(MTOT / 256, EE / 64), blk, 0, stream>>>(
        o_h, wo_h, b_o, x, x2, nullptr, EE, EE);

    // 6) LN2: x2 -> xn2 (f16)
    ln_kernel<<<MTOT / 8, blk, 0, stream>>>(x2, ln2_w, ln2_b, xn2_h);

    // 7) MLP1 + GELU: h = gelu(xn2*W1^T + b1) (f16 [M,FF])
    gemm_wmma_kernel<2><<<dim3(MTOT / 256, FFD / 64), blk, 0, stream>>>(
        xn2_h, w1_h, b1, nullptr, nullptr, h_h, FFD, EE);

    // 8) MLP2 + residual: out = x2 + h*W2^T + b2 (f32)
    gemm_wmma_kernel<1><<<dim3(MTOT / 256, EE / 64), blk, 0, stream>>>(
        h_h, w2_h, b2, x2, out, nullptr, EE, FFD);
}
